// MMGF_GCN_59055800320496
// MI455X (gfx1250) — matrix operand order repaired and checked by reading, not compiled, not run on hardware
//
#include <hip/hip_runtime.h>
#include <cstdint>
#include <cstddef>

// ---------------- problem constants ----------------
#define BB     128
#define NPRO   1442
#define MOLN   (BB*40)      // 5120
#define MOLE   (BB*160)     // 20480
#define PRON   (NPRO*100)   // 144200
#define PROE   500000
#define PPIE   20000
#define FPD    2048
#define HH     128

typedef __attribute__((ext_vector_type(16))) _Float16 v16h;
typedef __attribute__((ext_vector_type(8)))  _Float16 v8h;
typedef __attribute__((ext_vector_type(8)))  float    v8f;
typedef __attribute__((ext_vector_type(4)))  float    v4f;

#define WMMA_F16(a,b,c) __builtin_amdgcn_wmma_f32_16x16x32_f16(false,(a),false,(b),(short)0,(c),false,false)

// ---------------- elementwise / graph kernels ----------------
__global__ void k_fill(float* __restrict__ p, float v, size_t n){
  size_t i = (size_t)blockIdx.x*blockDim.x + threadIdx.x;
  if (i < n) p[i] = v;
}

__global__ void k_deg(const int* __restrict__ dst, int E, float* __restrict__ deg){
  int e = blockIdx.x*blockDim.x + threadIdx.x;
  if (e < E) atomicAdd(&deg[dst[e]], 1.0f);
}

__global__ void k_rsqrt(float* __restrict__ d, int n){
  int i = blockIdx.x*blockDim.x + threadIdx.x;
  if (i < n) d[i] = rsqrtf(d[i]);
}

// out[dst] += h[src] * dinv[src]*dinv[dst]
__global__ void k_scatter(const float* __restrict__ h, const int* __restrict__ src,
                          const int* __restrict__ dst, const float* __restrict__ dinv,
                          float* __restrict__ out, int E, int F){
  size_t idx = (size_t)blockIdx.x*blockDim.x + threadIdx.x;
  if (idx >= (size_t)E*F) return;
  int e = (int)(idx / F), f = (int)(idx % F);
  int s = src[e], d = dst[e];
  atomicAdd(&out[(size_t)d*F + f], h[(size_t)s*F + f] * dinv[s]*dinv[d]);
}

// out += h * dinv^2 (self loop) + bias, optional relu
__global__ void k_self(const float* __restrict__ h, const float* __restrict__ dinv,
                       const float* __restrict__ bias, float* __restrict__ out,
                       int Nn, int F, int relu){
  size_t idx = (size_t)blockIdx.x*blockDim.x + threadIdx.x;
  if (idx >= (size_t)Nn*F) return;
  int i = (int)(idx / F), f = (int)(idx % F);
  float v = out[idx] + h[idx]*dinv[i]*dinv[i] + bias[f];
  if (relu) v = fmaxf(v, 0.0f);
  out[idx] = v;
}

__global__ void k_pool_accum(const float* __restrict__ x, const int* __restrict__ batch,
                             float* __restrict__ sum, float* __restrict__ cnt, int Nn, int F){
  size_t idx = (size_t)blockIdx.x*blockDim.x + threadIdx.x;
  if (idx >= (size_t)Nn*F) return;
  int i = (int)(idx / F), f = (int)(idx % F);
  int s = batch[i];
  atomicAdd(&sum[(size_t)s*F + f], x[idx]);
  if (f == 0) atomicAdd(&cnt[s], 1.0f);
}

__global__ void k_pool_div(float* __restrict__ sum, const float* __restrict__ cnt, int S, int F){
  size_t idx = (size_t)blockIdx.x*blockDim.x + threadIdx.x;
  if (idx >= (size_t)S*F) return;
  int s = (int)(idx / F);
  sum[idx] = sum[idx] / fmaxf(cnt[s], 1.0f);
}

// out[i, 0:F] = a[idx[i]], out[i, F:2F] = b[idx[i]]
__global__ void k_gather2(const float* __restrict__ a, const float* __restrict__ b,
                          const int* __restrict__ idx, float* __restrict__ out, int Bn, int F){
  int t = blockIdx.x*blockDim.x + threadIdx.x;
  if (t >= Bn*2*F) return;
  int i = t / (2*F), f = t % (2*F);
  int r = idx[i];
  out[t] = (f < F) ? a[(size_t)r*F + f] : b[(size_t)r*F + (f - F)];
}

// pre-swizzle Wh (128x512 f32) into WMMA B-fragment order, f16
// frag = tileN*4 + kc ; element (lane, m): Wh[(kc*32 + (lane>>4)*16 + m)*512 + tileN*16 + (lane&15)]
__global__ void k_swz(const float* __restrict__ Wh, _Float16* __restrict__ out){
  int idx = blockIdx.x*blockDim.x + threadIdx.x;
  if (idx >= 128*512) return;
  int frag = idx >> 9;
  int lane = (idx >> 4) & 31;
  int m    = idx & 15;
  int tileN = frag >> 2, kc = frag & 3;
  int half = lane >> 4, l16 = lane & 15;
  int k   = kc*32 + half*16 + m;
  int col = tileN*16 + l16;
  out[idx] = (_Float16)Wh[k*512 + col];
}

// ---------------- generic WMMA GEMM: C[M,N](ldc) = A[M,K](lda) @ W[K,N](ldw) (+bias) (relu) ----
// 4 waves/block share one N-tile; B chunk is staged into LDS in fragment order once per block.
__global__ void __launch_bounds__(128) k_gemm(
    const float* __restrict__ A, int lda,
    const float* __restrict__ Wt, int ldw,
    const float* __restrict__ bias,
    float* __restrict__ C, int ldc,
    int M, int N, int K, int relu)
{
  __shared__ _Float16 bsh[512];      // 32x16 B chunk, fragment-ordered: [lane][m]
  int lane = threadIdx.x;            // 0..31 (one wave per 16x16 tile)
  int half = lane >> 4, l16 = lane & 15;
  int tid  = threadIdx.y * 32 + lane;
  int tileN = blockIdx.x;
  int tileM = blockIdx.y * blockDim.y + threadIdx.y;   // may exceed M range; stores guarded
  int row = tileM*16 + l16;
  int col = tileN*16 + l16;
  const bool alA = ((lda & 3) == 0) && ((((uintptr_t)A) & 15) == 0);
  const bool alW = ((ldw & 3) == 0) && ((((uintptr_t)Wt) & 15) == 0);
  const bool fullN = (tileN*16 + 16 <= N);
  // staging coords (fast path): one float4 of one W row per thread
  const int skk = tid >> 2;          // k within chunk (0..31)
  const int sc4 = (tid & 3) * 4;     // first of 4 consecutive columns
  const int sbase = (((skk >> 4)*16 + sc4) << 4) + (skk & 15);   // LDS frag index
  v8f acc = {};
  for (int k0 = 0; k0 < K; k0 += 32) {
    const bool fullK = (k0 + 32 <= K);
    // ---- cooperative, coalesced B staging (f32 -> f16, fragment order) ----
    if (fullK && fullN && alW) {
      // uniform fast path: one unguarded b128 load per thread
      v4f wv = *(const v4f*)(Wt + (size_t)(k0 + skk)*ldw + tileN*16 + sc4);
      bsh[sbase]      = (_Float16)wv[0];
      bsh[sbase + 16] = (_Float16)wv[1];
      bsh[sbase + 32] = (_Float16)wv[2];
      bsh[sbase + 48] = (_Float16)wv[3];
    } else {
      #pragma unroll
      for (int idx = 0; idx < 4; ++idx) {
        int e  = idx*128 + tid;        // 0..511
        int kk = e >> 4, cc = e & 15;  // kk: k within chunk, cc: col within tile
        int kg = k0 + kk, cg = tileN*16 + cc;
        float v = 0.0f;
        if (kg < K && cg < N) v = Wt[(size_t)kg*ldw + cg];
        bsh[(((kk >> 4)*16 + cc) << 4) + (kk & 15)] = (_Float16)v;
      }
    }
    __syncthreads();
    v16h b = *(const v16h*)(bsh + lane*16);
    // ---- A fragment ----
    v16h a = {};
    if (row < M) {
      const float* ar = A + (size_t)row*lda;
      if (fullK && alA) {
        v4f f0 = *(const v4f*)(ar + k0 + half*16);
        v4f f1 = *(const v4f*)(ar + k0 + half*16 + 4);
        v4f f2 = *(const v4f*)(ar + k0 + half*16 + 8);
        v4f f3 = *(const v4f*)(ar + k0 + half*16 + 12);
        #pragma unroll
        for (int q = 0; q < 4; ++q) {
          a[q]    = (_Float16)f0[q];
          a[4+q]  = (_Float16)f1[q];
          a[8+q]  = (_Float16)f2[q];
          a[12+q] = (_Float16)f3[q];
        }
      } else if (fullK) {
        #pragma unroll
        for (int q = 0; q < 8; ++q) {
          a[q]   = (_Float16)ar[k0 + half*16 + q];
          a[8+q] = (_Float16)ar[k0 + half*16 + 8 + q];
        }
      } else {
        #pragma unroll
        for (int q = 0; q < 8; ++q) {
          int k1 = k0 + half*16 + q;
          if (k1 < K) a[q]   = (_Float16)ar[k1];
          int k2 = k0 + half*16 + 8 + q;
          if (k2 < K) a[8+q] = (_Float16)ar[k2];
        }
      }
    }
    acc = WMMA_F16(a, b, acc);
    __syncthreads();                 // protect bsh before next chunk's staging
  }
  #pragma unroll
  for (int r = 0; r < 8; ++r) {
    int m = tileM*16 + half*8 + r;
    if (m < M && col < N) {
      float v = acc[r];
      if (bias) v += bias[col];
      if (relu) v = fmaxf(v, 0.0f);
      C[(size_t)m*ldc + col] = v;
    }
  }
}

// ---------------- persistent single-workgroup WMMA LSTM ----------------
// fp:[128,2048] f32, Wi:[512], whswz: pre-swizzled Wh f16 (65536), b:[512], hout:[128,128] f32
__global__ void __launch_bounds__(512, 1) k_lstm(
    const float* __restrict__ fp, const float* __restrict__ Wi,
    const _Float16* __restrict__ whswz, const float* __restrict__ bg,
    float* __restrict__ hout)
{
  extern __shared__ _Float16 lsmem[];
  _Float16* whlds = lsmem;           // 65536 f16 (128 KB)
  _Float16* hl    = lsmem + 65536;   // 16384 f16 (32 KB)
  int tid = threadIdx.x;
  for (int i = tid; i < 65536; i += 512) whlds[i] = whswz[i];
  for (int i = tid; i < 16384; i += 512) hl[i] = (_Float16)0.0f;

  int wv = tid >> 5, lane = tid & 31;
  int half = lane >> 4, l16 = lane & 15;
  int tm = wv & 7;                   // row tile (rows tm*16 .. +15)
  int gbase = (wv >> 3) * 4;         // hidden-col group base (4 groups per wave)
  int rbase = tm*16 + half*8;

  float wiv[4][4], bbv[4][4];
  #pragma unroll
  for (int j = 0; j < 4; ++j)
    #pragma unroll
    for (int p = 0; p < 4; ++p) {
      int cc = p*128 + (gbase + j)*16 + l16;
      wiv[j][p] = Wi[cc];
      bbv[j][p] = bg[cc];
    }
  v8f cst[4] = {};
  __syncthreads();

  int arow = tm*16 + l16;
  for (int t = 0; t < FPD; ++t) {
    // load A fragments (h) from LDS — the only h reads this step
    v16h afr[4];
    #pragma unroll
    for (int kc = 0; kc < 4; ++kc) {
      const _Float16* base = hl + arow*128 + kc*32;
      v8h lo = *(const v8h*)(base + half*16);
      v8h hi = *(const v8h*)(base + half*16 + 8);
      v16h a;
      #pragma unroll
      for (int i = 0; i < 8; ++i) { a[i] = lo[i]; a[8+i] = hi[i]; }
      afr[kc] = a;
    }
    __syncthreads();   // all reads done before anyone rewrites h

    v8f acc[4][4];
    #pragma unroll
    for (int j = 0; j < 4; ++j)
      #pragma unroll
      for (int p = 0; p < 4; ++p) {
        int tn = p*8 + gbase + j;
        v8f ac = {};
        #pragma unroll
        for (int kc = 0; kc < 4; ++kc) {
          v16h bfr = *(const v16h*)(whlds + ((size_t)(tn*4 + kc)*32 + lane)*16);
          ac = WMMA_F16(afr[kc], bfr, ac);
        }
        acc[j][p] = ac;
      }

    #pragma unroll
    for (int r = 0; r < 8; ++r) {
      float xt = fp[(size_t)(rbase + r)*FPD + t];
      #pragma unroll
      for (int j = 0; j < 4; ++j) {
        float gi = acc[j][0][r] + xt*wiv[j][0] + bbv[j][0];
        float gf = acc[j][1][r] + xt*wiv[j][1] + bbv[j][1];
        float gg = acc[j][2][r] + xt*wiv[j][2] + bbv[j][2];
        float go = acc[j][3][r] + xt*wiv[j][3] + bbv[j][3];
        float ii = 1.0f/(1.0f + __expf(-gi));
        float ff = 1.0f/(1.0f + __expf(-gf));
        float oo = 1.0f/(1.0f + __expf(-go));
        float cv = ff*cst[j][r] + ii*tanhf(gg);
        cst[j][r] = cv;
        float hv = oo*tanhf(cv);
        int hidx = (rbase + r)*128 + (gbase + j)*16 + l16;
        hl[hidx] = (_Float16)hv;
        if (t == FPD-1) hout[hidx] = hv;
      }
    }
    __syncthreads();   // h writes visible before next step's reads
  }
}

// ---------------- host orchestration ----------------
extern "C" void kernel_launch(void* const* d_in, const int* in_sizes, int n_in,
                              void* d_out, int out_size, void* d_ws, size_t ws_size,
                              hipStream_t stream) {
  (void)out_size; (void)ws_size;
  auto F32 = [&](int i)->const float* { return (const float*)d_in[i]; };
  const float* mol_x  = F32(0);
  const int*   mol_ei = (const int*)d_in[1];
  const int*   mol_b  = (const int*)d_in[2];
  const int*   seqn   = (const int*)d_in[3];
  const int*   ppi_ei = (const int*)d_in[4];
  const float* ppi_x  = F32(5);
  const float* pro_x  = F32(6);
  const int*   pro_ei = (const int*)d_in[7];
  const int*   pro_b  = (const int*)d_in[8];
  const float* fp_t   = F32(9);

  // -------- param leaves: handle insertion-order vs jax-sorted pytree order --------
  const float *mgW[4],*mgB[4],*molFC1W,*molFC1B,*molFC2W,*molFC2B;
  const float *pgW[3],*pgB[3],*proFC1W,*proFC1B,*proFC2W,*proFC2B;
  const float *qgW[4],*qgB[4],*ppiFC1W,*ppiFC1B,*ppiFC2W,*ppiFC2B;
  const float *pcW,*pcB,*WiP,*WhP,*lbP,*fpfcW,*fpfcB;
  const float *WvpW,*WvpB,*WvxW,*WvxB,*fcxW,*fcxB,*fcpW,*fcpB;
  const float *fc1W,*fc1B,*fc2W,*fc2B,*outW,*outB;
  bool insertion = (n_in > 10) && (in_sizes[10] == 78*156);
  int i = 10;
  if (insertion) {
    for (int l=0;l<4;++l){ mgW[l]=F32(i++); mgB[l]=F32(i++); }
    molFC1W=F32(i++); molFC1B=F32(i++); molFC2W=F32(i++); molFC2B=F32(i++);
    for (int l=0;l<3;++l){ pgW[l]=F32(i++); pgB[l]=F32(i++); }
    proFC1W=F32(i++); proFC1B=F32(i++); proFC2W=F32(i++); proFC2B=F32(i++);
    for (int l=0;l<4;++l){ qgW[l]=F32(i++); qgB[l]=F32(i++); }
    ppiFC1W=F32(i++); ppiFC1B=F32(i++); ppiFC2W=F32(i++); ppiFC2B=F32(i++);
    pcW=F32(i++); pcB=F32(i++);
    WiP=F32(i++); WhP=F32(i++); lbP=F32(i++);
    fpfcW=F32(i++); fpfcB=F32(i++);
    i += 2;                               // Wqx
    i += 2;                               // Wkp
    WvpW=F32(i++); WvpB=F32(i++);
    i += 2;                               // Wqp
    i += 2;                               // Wkx
    WvxW=F32(i++); WvxB=F32(i++);
    fcxW=F32(i++); fcxB=F32(i++);
    fcpW=F32(i++); fcpB=F32(i++);
    fc1W=F32(i++); fc1B=F32(i++); fc2W=F32(i++); fc2B=F32(i++);
    outW=F32(i++); outB=F32(i++);
  } else {                                // sorted pytree order
    i += 2;                               // Wkp
    i += 2;                               // Wkx
    i += 2;                               // Wqp
    i += 2;                               // Wqx
    WvpW=F32(i++); WvpB=F32(i++);
    WvxW=F32(i++); WvxB=F32(i++);
    fcpW=F32(i++); fcpB=F32(i++);
    fcxW=F32(i++); fcxB=F32(i++);
    fc1W=F32(i++); fc1B=F32(i++);
    fc2W=F32(i++); fc2B=F32(i++);
    fpfcW=F32(i++); fpfcB=F32(i++);
    WhP=F32(i++); WiP=F32(i++); lbP=F32(i++);
    molFC1W=F32(i++); molFC1B=F32(i++); molFC2W=F32(i++); molFC2B=F32(i++);
    for (int l=0;l<4;++l){ mgW[l]=F32(i++); mgB[l]=F32(i++); }
    outW=F32(i++); outB=F32(i++);
    ppiFC1W=F32(i++); ppiFC1B=F32(i++); ppiFC2W=F32(i++); ppiFC2B=F32(i++);
    for (int l=0;l<4;++l){ qgW[l]=F32(i++); qgB[l]=F32(i++); }
    proFC1W=F32(i++); proFC1B=F32(i++); proFC2W=F32(i++); proFC2B=F32(i++);
    pcW=F32(i++); pcB=F32(i++);
    for (int l=0;l<3;++l){ pgW[l]=F32(i++); pgB[l]=F32(i++); }
  }

  // -------- workspace carving --------
  char* wsb = (char*)d_ws; size_t off = 0;
  auto alloc = [&](size_t bytes)->void* {
    void* p = wsb + off; off = (off + bytes + 255) & ~(size_t)255; return p;
  };
  float* bufA = (float*)alloc((size_t)PRON*128*4);
  float* bufB = (float*)alloc((size_t)PRON*128*4);
  float* bufC = (float*)alloc((size_t)PRON*128*4);
  float* S1   = (float*)alloc((size_t)NPRO*1024*4);
  float* deg  = (float*)alloc((size_t)PRON*4);
  float* cnt  = (float*)alloc((size_t)NPRO*4);
  float* molP = (float*)alloc((size_t)BB*128*4);
  float* proP = (float*)alloc((size_t)NPRO*128*4);
  float* drug = (float*)alloc((size_t)BB*256*4);
  float* proOne = (float*)alloc((size_t)NPRO*128*4);
  float* ppiDual= (float*)alloc((size_t)NPRO*128*4);
  float* proCat = (float*)alloc((size_t)BB*256*4);
  float* proFeat= (float*)alloc((size_t)BB*128*4);
  float* Vx   = (float*)alloc((size_t)BB*64*4);
  float* Vp   = (float*)alloc((size_t)BB*64*4);
  float* comb = (float*)alloc((size_t)BB*384*4);
  float* f2o  = (float*)alloc((size_t)BB*512*4);
  _Float16* whswz = (_Float16*)alloc((size_t)128*512*2);
  float* hfin = (float*)alloc((size_t)BB*128*4);

  // -------- launch helpers --------
  auto fill = [&](float* p, float v, size_t n){
    k_fill<<<dim3((unsigned)((n+255)/256)),dim3(256),0,stream>>>(p,v,n);
  };
  auto gemm = [&](const float* A,int lda,const float* Wm,int Nc,const float* bias,
                  float* C,int ldc,int M,int K,int relu){
    dim3 blk(32,4), grd((Nc+15)/16, (M+63)/64);
    k_gemm<<<grd,blk,0,stream>>>(A,lda,Wm,Nc,bias,C,ldc,M,Nc,K,relu);
  };
  auto gcn = [&](const float* x,int K,int Fo,const float* Wl,const float* bl,
                 const int* src,const int* dst,const float* dv,int Nn,int E,
                 float* htmp,float* out){
    gemm(x,K,Wl,Fo,nullptr,htmp,Fo,Nn,K,0);
    fill(out,0.0f,(size_t)Nn*Fo);
    size_t tot=(size_t)E*Fo;
    k_scatter<<<dim3((unsigned)((tot+255)/256)),dim3(256),0,stream>>>(htmp,src,dst,dv,out,E,Fo);
    size_t t2=(size_t)Nn*Fo;
    k_self<<<dim3((unsigned)((t2+255)/256)),dim3(256),0,stream>>>(htmp,dv,bl,out,Nn,Fo,1);
  };
  auto mkdeg = [&](const int* dst,int E,int Nn){
    fill(deg,1.0f,(size_t)Nn);
    k_deg<<<dim3((E+255)/256),dim3(256),0,stream>>>(dst,E,deg);
    k_rsqrt<<<dim3((Nn+255)/256),dim3(256),0,stream>>>(deg,Nn);
  };
  auto pool = [&](const float* x,const int* batch,int Nn,int F,int S,float* out){
    fill(out,0.0f,(size_t)S*F); fill(cnt,0.0f,(size_t)S);
    size_t tot=(size_t)Nn*F;
    k_pool_accum<<<dim3((unsigned)((tot+255)/256)),dim3(256),0,stream>>>(x,batch,out,cnt,Nn,F);
    size_t t2=(size_t)S*F;
    k_pool_div<<<dim3((unsigned)((t2+255)/256)),dim3(256),0,stream>>>(out,cnt,S,F);
  };

  // ======== molecular graph branch ========
  mkdeg(mol_ei + MOLE, MOLE, MOLN);
  gcn(mol_x,78,156, mgW[0],mgB[0], mol_ei,mol_ei+MOLE,deg, MOLN,MOLE, bufA,bufB);
  gcn(bufB,156,312, mgW[1],mgB[1], mol_ei,mol_ei+MOLE,deg, MOLN,MOLE, bufA,bufC);
  gcn(bufC,312,128, mgW[2],mgB[2], mol_ei,mol_ei+MOLE,deg, MOLN,MOLE, bufA,bufB);
  gcn(bufB,128,128, mgW[3],mgB[3], mol_ei,mol_ei+MOLE,deg, MOLN,MOLE, bufA,bufC);
  pool(bufC, mol_b, MOLN, 128, BB, molP);
  gemm(molP,128, molFC1W,1024, molFC1B, S1,1024, BB,128, 1);
  gemm(S1,1024,  molFC2W,128,  molFC2B, drug,256, BB,1024, 0);     // drug[:,0:128]

  // ======== fingerprint LSTM branch ========
  k_swz<<<dim3((128*512+255)/256),dim3(256),0,stream>>>(WhP, whswz);
  k_lstm<<<dim3(1),dim3(512), (65536+16384)*sizeof(_Float16), stream>>>(fp_t, WiP, whswz, lbP, hfin);
  gemm(hfin,128, fpfcW,128, fpfcB, drug+128,256, BB,128, 0);       // drug[:,128:256]

  // ======== protein residue-graph branch ========
  mkdeg(pro_ei + PROE, PROE, PRON);
  gcn(pro_x,33,128,  pgW[0],pgB[0], pro_ei,pro_ei+PROE,deg, PRON,PROE, bufA,bufB);
  gcn(bufB,128,128,  pgW[1],pgB[1], pro_ei,pro_ei+PROE,deg, PRON,PROE, bufA,bufC);
  gcn(bufC,128,128,  pgW[2],pgB[2], pro_ei,pro_ei+PROE,deg, PRON,PROE, bufA,bufB);
  pool(bufB, pro_b, PRON, 128, NPRO, proP);
  gemm(proP,128, proFC1W,1024, proFC1B, S1,1024, NPRO,128, 1);
  gemm(S1,1024,  proFC2W,128,  proFC2B, proOne,128, NPRO,1024, 0);

  // ======== PPI graph branch ========
  mkdeg(ppi_ei + PPIE, PPIE, NPRO);
  gcn(ppi_x,NPRO,1024, qgW[0],qgB[0], ppi_ei,ppi_ei+PPIE,deg, NPRO,PPIE, bufA,bufB);
  gcn(bufB,1024,512,   qgW[1],qgB[1], ppi_ei,ppi_ei+PPIE,deg, NPRO,PPIE, bufA,bufC);
  gcn(bufC,512,128,    qgW[2],qgB[2], ppi_ei,ppi_ei+PPIE,deg, NPRO,PPIE, bufA,bufB);
  gcn(bufB,128,128,    qgW[3],qgB[3], ppi_ei,ppi_ei+PPIE,deg, NPRO,PPIE, bufA,bufC);
  gemm(bufC,128, ppiFC1W,1024, ppiFC1B, S1,1024, NPRO,128, 1);
  gemm(S1,1024,  ppiFC2W,128,  ppiFC2B, ppiDual,128, NPRO,1024, 0);

  // ======== combine ========
  k_gather2<<<dim3((BB*256+255)/256),dim3(256),0,stream>>>(proOne, ppiDual, seqn, proCat, BB, 128);
  gemm(proCat,256, pcW,128, pcB, proFeat,128, BB,256, 0);

  // co-attention: softmax over singleton axis == 1 -> attended = fc(V)
  gemm(drug,256,    WvxW,64,  WvxB, Vx,64, BB,256, 0);
  gemm(Vx,64,       fcxW,256, fcxB, comb,384, BB,64, 0);           // comb[:,0:256]
  gemm(proFeat,128, WvpW,64,  WvpB, Vp,64, BB,128, 0);
  gemm(Vp,64,       fcpW,128, fcpB, comb+256,384, BB,64, 0);       // comb[:,256:384]

  gemm(comb,384, fc1W,1024, fc1B, S1,1024, BB,384, 1);
  gemm(S1,1024,  fc2W,512,  fc2B, f2o,512, BB,1024, 1);
  gemm(f2o,512,  outW,1,    outB, (float*)d_out,1, BB,512, 0);
}
